// PAMoE_53042846105701
// MI455X (gfx1250) — matrix operand order repaired and checked by reading, not compiled, not run on hardware
//
#include <hip/hip_runtime.h>
#include <hip/hip_bf16.h>
#include <math.h>

// Problem constants (match reference)
#define Bc   4
#define Tc   8192
#define Dc   256
#define Ec   8
#define FFNc 512
#define TOPK 1024
#define EPSc 1e-5f

// packed-weight tile geometry
#define NT1  (FFNc / 16)   // 32 n-tiles for fc1
#define KS1  (Dc / 32)     // 8  k-steps for fc1
#define NT2  (Dc / 16)     // 16 n-tiles for fc2
#define KS2  (FFNc / 32)   // 16 k-steps for fc2
#define W1P_ELEMS ((size_t)Ec * FFNc * Dc)   // f16 elements
#define W2P_ELEMS ((size_t)Ec * Dc * FFNc)

typedef __attribute__((ext_vector_type(16))) _Float16 v16h;
typedef __attribute__((ext_vector_type(8)))  _Float16 v8h;
typedef __attribute__((ext_vector_type(8)))  float    v8f;

// ---------------------------------------------------------------------------
// 0) zero the scatter-add output region
// ---------------------------------------------------------------------------
__global__ void pamoe_zero_kernel(float* __restrict__ p, int n4) {
    int i = blockIdx.x * blockDim.x + threadIdx.x;
    if (i < n4) {
        float4 z; z.x = 0.f; z.y = 0.f; z.z = 0.f; z.w = 0.f;
        ((float4*)p)[i] = z;
    }
}

// ---------------------------------------------------------------------------
// 0b) pack weights fp32 -> f16 in WMMA-B fragment order:
//     [e][nTile][kStep][lane(0..31)][j(0..15)], B element (k=kb+8*(lane/16)+(j&7)+16*(j/8), n=nB+lane%16).
// ---------------------------------------------------------------------------
__global__ void pamoe_pack_w_kernel(const float* __restrict__ W, // [E, N, K]
                                    _Float16* __restrict__ out,
                                    int Ndim, int Kdim, int nTiles, int kSteps) {
    int idx = blockIdx.x * blockDim.x + threadIdx.x;
    int total = Ec * nTiles * kSteps * 32;
    if (idx >= total) return;
    int lane = idx & 31;
    int t = idx >> 5;
    int ks = t % kSteps; t /= kSteps;
    int nt = t % nTiles; t /= nTiles;
    int e  = t;
    const float* src = W + ((size_t)e * Ndim + nt * 16 + (lane & 15)) * Kdim + ks * 32 + 8 * (lane >> 4);
    _Float16* dst = out + (size_t)idx * 16;
    #pragma unroll
    for (int j = 0; j < 16; ++j)
        dst[j] = (_Float16)src[(j & 7) + ((j >> 3) << 4)];
}

// ---------------------------------------------------------------------------
// 1) router: logits = x @ Wr^T, softmax over E.
// ---------------------------------------------------------------------------
__global__ void pamoe_router_kernel(const float* __restrict__ x,
                                    const float* __restrict__ Wr,
                                    float* __restrict__ logits,
                                    float* __restrict__ probs) {
    int idx = blockIdx.x * blockDim.x + threadIdx.x;   // b*T + t
    if (idx >= Bc * Tc) return;
    int b = idx / Tc, t = idx % Tc;

    float acc[Ec];
    #pragma unroll
    for (int e = 0; e < Ec; ++e) acc[e] = 0.f;

    const float* row = x + (size_t)idx * Dc;
    for (int d = 0; d < Dc; ++d) {
        float xv = row[d];
        #pragma unroll
        for (int e = 0; e < Ec; ++e) acc[e] += xv * Wr[e * Dc + d];
    }

    float mx = acc[0];
    #pragma unroll
    for (int e = 1; e < Ec; ++e) mx = fmaxf(mx, acc[e]);
    float sum = 0.f;
    float ex[Ec];
    #pragma unroll
    for (int e = 0; e < Ec; ++e) { ex[e] = __expf(acc[e] - mx); sum += ex[e]; }
    float inv = 1.f / sum;

    #pragma unroll
    for (int e = 0; e < Ec; ++e) {
        logits[(size_t)idx * Ec + e] = acc[e];
        probs[((size_t)b * Ec + e) * Tc + t] = ex[e] * inv;
    }
}

// ---------------------------------------------------------------------------
// 2) expert-choice top-k: per (b,e) bitonic sort in LDS, keep top 1024.
// ---------------------------------------------------------------------------
__global__ __launch_bounds__(256) void pamoe_topk_kernel(
        const float* __restrict__ probs,
        int* __restrict__ sel, float* __restrict__ wval) {
    __shared__ unsigned long long keys[Tc];   // 64 KB of the 320 KB WGP LDS
    const int be = blockIdx.x;                // b*E + e
    const float* p = probs + (size_t)be * Tc;

    for (int i = threadIdx.x; i < Tc; i += blockDim.x) {
        unsigned u = __float_as_uint(p[i]);
        unsigned k32 = (u & 0x80000000u) ? ~u : (u | 0x80000000u); // total order
        keys[i] = ((unsigned long long)k32 << 32) | (unsigned)(~(unsigned)i);
    }
    __syncthreads();

    for (int kk = 2; kk <= Tc; kk <<= 1) {
        for (int jj = kk >> 1; jj > 0; jj >>= 1) {
            for (int i = threadIdx.x; i < Tc; i += blockDim.x) {
                int l = i ^ jj;
                if (l > i) {
                    unsigned long long a = keys[i], c = keys[l];
                    if ((a < c) == ((i & kk) == 0)) { keys[i] = c; keys[l] = a; }
                }
            }
            __syncthreads();
        }
    }

    for (int i = threadIdx.x; i < TOPK; i += blockDim.x) {
        unsigned long long kv = keys[i];
        unsigned k32 = (unsigned)(kv >> 32);
        unsigned u = (k32 & 0x80000000u) ? (k32 & 0x7FFFFFFFu) : ~k32;
        sel[(size_t)be * TOPK + i]  = (int)~((unsigned)kv);
        wval[(size_t)be * TOPK + i] = __uint_as_float(u);
    }
}

// ---------------------------------------------------------------------------
// fragment load helpers for the FFN kernel
// ---------------------------------------------------------------------------
__device__ __forceinline__ v16h load_a_frag(const _Float16* lds, int ln, int g,
                                            int stride, int kb) {
    v8h alo = *(const v8h*)&lds[ln * stride + kb + 8 * g];
    v8h ahi = *(const v8h*)&lds[ln * stride + kb + 16 + 8 * g];
    v16h a;
    #pragma unroll
    for (int i = 0; i < 8; ++i) { a[i] = alo[i]; a[8 + i] = ahi[i]; }
    return a;
}

__device__ __forceinline__ v16h load_b_frag(const _Float16* We, int nTile,
                                            int kSteps, int ks, int lane) {
    return *(const v16h*)(We + (((size_t)nTile * kSteps + ks) * 32 + lane) * 16);
}

// ---------------------------------------------------------------------------
// 3) MoE FFN with WMMA: ping-pong double-buffered fragments (no reg moves).
//    One workgroup = 8 wave32 = one (e, b, 16-token tile).
// ---------------------------------------------------------------------------
__global__ __launch_bounds__(256) void pamoe_ffn_kernel(
        const float* __restrict__ x,
        const _Float16* __restrict__ W1p, const float* __restrict__ b1,
        const float* __restrict__ gamma, const float* __restrict__ beta,
        const _Float16* __restrict__ W2p, const float* __restrict__ b2,
        const int* __restrict__ sel, const float* __restrict__ wval,
        float* __restrict__ out) {
    __shared__ __align__(16) _Float16 ldsX[16 * 264];  // A tile for fc1 (f16)
    __shared__ __align__(16) _Float16 ldsH[16 * 520];  // A tile for fc2 (f16)
    __shared__ float hbuf[16 * 516];                   // fc1 activations (f32)
    __shared__ float psum[16 * 16], psq[16 * 16];
    __shared__ float rowMu[16], rowRs[16];
    __shared__ int   selInS[16], selOutS[16];
    __shared__ float wOutS[16];

    const int tid  = threadIdx.x;
    const int lane = tid & 31;
    const int w    = tid >> 5;          // wave 0..7
    const int g    = lane >> 4;         // half-wave group
    const int ln   = lane & 15;
    const int mBase = blockIdx.x * 16;  // token tile inside TOPK
    const int b = blockIdx.y;
    const int e = blockIdx.z;
    const int flat = e * Bc + b;        // reference reshape quirk
    const int bp = flat / Ec;
    const int ep = flat % Ec;

    if (tid < 16) {
        selInS[tid]  = sel[((size_t)bp * Ec + ep) * TOPK + mBase + tid];
        selOutS[tid] = sel[((size_t)b  * Ec + e ) * TOPK + mBase + tid];
        wOutS[tid]   = wval[((size_t)b * Ec + e ) * TOPK + mBase + tid];
    }
    __syncthreads();

    // Gather 16 token rows, f32 -> f16 into LDS (row-major, stride 264)
    {
        int row = tid >> 4;
        int c0  = (tid & 15) * 16;
        const float* src = x + ((size_t)bp * Tc + selInS[row]) * Dc + c0;
        #pragma unroll
        for (int c = 0; c < 16; ++c)
            ldsX[row * 264 + c0 + c] = (_Float16)src[c];
    }
    __syncthreads();

    // ---------------- FC1: each wave owns 64 FFN columns ----------------
    v8f acc[4] = {};
    const _Float16* W1e = W1p + ((size_t)e * NT1 * KS1) * 512; // 512 f16/tile
    {
        v16h aF[2];
        v16h bF[2][4];
        aF[0] = load_a_frag(ldsX, ln, g, 264, 0);
        #pragma unroll
        for (int nt = 0; nt < 4; ++nt)
            bF[0][nt] = load_b_frag(W1e, w * 4 + nt, KS1, 0, lane);

        #pragma unroll 1
        for (int ks = 0; ks < KS1; ks += 2) {          // KS1 even
            const int k1 = (ks + 1 < KS1) ? ks + 1 : ks;
            aF[1] = load_a_frag(ldsX, ln, g, 264, k1 * 32);
            #pragma unroll
            for (int nt = 0; nt < 4; ++nt)
                bF[1][nt] = load_b_frag(W1e, w * 4 + nt, KS1, k1, lane);
            if (ks + 4 < KS1)   // stream K-slab ks+4 -> global_prefetch_b8
                __builtin_prefetch(
                    W1e + (((size_t)(w * 4) * KS1 + ks + 4) * 32 + lane) * 16, 0, 1);
            #pragma unroll
            for (int nt = 0; nt < 4; ++nt)
                acc[nt] = __builtin_amdgcn_wmma_f32_16x16x32_f16(
                    false, aF[0], false, bF[0][nt], (short)0, acc[nt], false, false);

            const int k2 = (ks + 2 < KS1) ? ks + 2 : k1;
            aF[0] = load_a_frag(ldsX, ln, g, 264, k2 * 32);
            #pragma unroll
            for (int nt = 0; nt < 4; ++nt)
                bF[0][nt] = load_b_frag(W1e, w * 4 + nt, KS1, k2, lane);
            #pragma unroll
            for (int nt = 0; nt < 4; ++nt)
                acc[nt] = __builtin_amdgcn_wmma_f32_16x16x32_f16(
                    false, aF[1], false, bF[1][nt], (short)0, acc[nt], false, false);
        }
    }

    // bias + exact GELU, spill to LDS for the cross-lane LayerNorm
    #pragma unroll
    for (int nt = 0; nt < 4; ++nt) {
        const int n = w * 64 + nt * 16 + ln;
        const float bias = b1[e * FFNc + n];
        #pragma unroll
        for (int r = 0; r < 8; ++r) {
            float h = acc[nt][r] + bias;
            float ge = 0.5f * h * (1.0f + erff(h * 0.70710678118654752f));
            hbuf[(r + 8 * g) * 516 + n] = ge;
        }
    }
    __syncthreads();

    // LayerNorm stats (biased variance, matches torch LayerNorm)
    {
        int row = tid >> 4, seg = tid & 15;
        float s = 0.f, sq = 0.f;
        const float* hr = &hbuf[row * 516 + seg * 32];
        #pragma unroll
        for (int c = 0; c < 32; ++c) { float v = hr[c]; s += v; sq += v * v; }
        psum[row * 16 + seg] = s; psq[row * 16 + seg] = sq;
    }
    __syncthreads();
    if (tid < 16) {
        float s = 0.f, sq = 0.f;
        #pragma unroll
        for (int seg = 0; seg < 16; ++seg) { s += psum[tid * 16 + seg]; sq += psq[tid * 16 + seg]; }
        float mu  = s / (float)FFNc;
        float var = sq / (float)FFNc - mu * mu;
        rowMu[tid] = mu;
        rowRs[tid] = rsqrtf(var + EPSc);
    }
    __syncthreads();
    {
        int row = tid >> 4, c0 = (tid & 15) * 32;
        float mu = rowMu[row], rs = rowRs[row];
        #pragma unroll
        for (int c = 0; c < 32; ++c) {
            int col = c0 + c;
            float v = (hbuf[row * 516 + col] - mu) * rs * gamma[e * FFNc + col]
                      + beta[e * FFNc + col];
            ldsH[row * 520 + col] = (_Float16)v;
        }
    }
    __syncthreads();

    // ---------------- FC2: each wave owns 32 D columns ----------------
    v8f acc2[2] = {};
    const _Float16* W2e = W2p + ((size_t)e * NT2 * KS2) * 512;
    {
        v16h aF[2];
        v16h bF[2][2];
        aF[0] = load_a_frag(ldsH, ln, g, 520, 0);
        #pragma unroll
        for (int nt = 0; nt < 2; ++nt)
            bF[0][nt] = load_b_frag(W2e, w * 2 + nt, KS2, 0, lane);

        #pragma unroll 1
        for (int ks = 0; ks < KS2; ks += 2) {          // KS2 even
            const int k1 = (ks + 1 < KS2) ? ks + 1 : ks;
            aF[1] = load_a_frag(ldsH, ln, g, 520, k1 * 32);
            #pragma unroll
            for (int nt = 0; nt < 2; ++nt)
                bF[1][nt] = load_b_frag(W2e, w * 2 + nt, KS2, k1, lane);
            if (ks + 4 < KS2)
                __builtin_prefetch(
                    W2e + (((size_t)(w * 2) * KS2 + ks + 4) * 32 + lane) * 16, 0, 1);
            #pragma unroll
            for (int nt = 0; nt < 2; ++nt)
                acc2[nt] = __builtin_amdgcn_wmma_f32_16x16x32_f16(
                    false, aF[0], false, bF[0][nt], (short)0, acc2[nt], false, false);

            const int k2 = (ks + 2 < KS2) ? ks + 2 : k1;
            aF[0] = load_a_frag(ldsH, ln, g, 520, k2 * 32);
            #pragma unroll
            for (int nt = 0; nt < 2; ++nt)
                bF[0][nt] = load_b_frag(W2e, w * 2 + nt, KS2, k2, lane);
            #pragma unroll
            for (int nt = 0; nt < 2; ++nt)
                acc2[nt] = __builtin_amdgcn_wmma_f32_16x16x32_f16(
                    false, aF[1], false, bF[1][nt], (short)0, acc2[nt], false, false);
        }
    }

    // bias, router-weight scale, scatter-add into out[b, t, :]
    #pragma unroll
    for (int nt = 0; nt < 2; ++nt) {
        const int n = w * 32 + nt * 16 + ln;
        const float bias = b2[e * Dc + n];
        #pragma unroll
        for (int r = 0; r < 8; ++r) {
            int m = r + 8 * g;
            float y = (acc2[nt][r] + bias) * wOutS[m];
            atomicAdd(&out[((size_t)b * Tc + selOutS[m]) * Dc + n], y);
        }
    }
}

// ---------------------------------------------------------------------------
extern "C" void kernel_launch(void* const* d_in, const int* in_sizes, int n_in,
                              void* d_out, int out_size, void* d_ws, size_t ws_size,
                              hipStream_t stream) {
    const float* x     = (const float*)d_in[0];
    const float* Wr    = (const float*)d_in[1];
    const float* W1    = (const float*)d_in[2];
    const float* b1    = (const float*)d_in[3];
    const float* gamma = (const float*)d_in[4];
    const float* beta  = (const float*)d_in[5];
    const float* W2    = (const float*)d_in[6];
    const float* b2    = (const float*)d_in[7];

    float* out    = (float*)d_out;
    float* logits = out + (size_t)Bc * Tc * Dc;       // tuple output #2

    float*    probs = (float*)d_ws;                          // [B,E,T]
    int*      sel   = (int*)(probs + (size_t)Bc * Ec * Tc);  // [B,E,TOPK]
    float*    wv    = (float*)(sel + (size_t)Bc * Ec * TOPK);// [B,E,TOPK]
    _Float16* W1pk  = (_Float16*)(wv + (size_t)Bc * Ec * TOPK);
    _Float16* W2pk  = W1pk + W1P_ELEMS;

    int n4 = (Bc * Tc * Dc) / 4;
    pamoe_zero_kernel<<<(n4 + 255) / 256, 256, 0, stream>>>(out, n4);

    int nPack1 = Ec * NT1 * KS1 * 32;   // threads for W1 pack
    int nPack2 = Ec * NT2 * KS2 * 32;   // threads for W2 pack
    pamoe_pack_w_kernel<<<(nPack1 + 255) / 256, 256, 0, stream>>>(
        W1, W1pk, FFNc, Dc, NT1, KS1);
    pamoe_pack_w_kernel<<<(nPack2 + 255) / 256, 256, 0, stream>>>(
        W2, W2pk, Dc, FFNc, NT2, KS2);

    pamoe_router_kernel<<<(Bc * Tc + 255) / 256, 256, 0, stream>>>(x, Wr, logits, probs);
    pamoe_topk_kernel<<<Bc * Ec, 256, 0, stream>>>(probs, sel, wv);
    pamoe_ffn_kernel<<<dim3(TOPK / 16, Bc, Ec), 256, 0, stream>>>(
        x, W1pk, b1, gamma, beta, W2pk, b2, sel, wv, out);
}
